// TopNGating_37623913513615
// MI455X (gfx1250) — matrix operand order repaired and checked by reading, not compiled, not run on hardware
//
#include <hip/hip_runtime.h>
#include <hip/hip_bf16.h>

typedef _Float16 v16h __attribute__((ext_vector_type(16)));
typedef float    v8f  __attribute__((ext_vector_type(8)));

#define B_    4
#define N_    2048
#define DIM_  1024
#define NE    8
#define NTOK  (B_ * N_)          // 8192
#define CAP   320
#define OUT_MAT ((size_t)NTOK * NE * CAP)   // 20,971,520 per tensor

// workspace layout (units: 4-byte words)
#define WS_LOG   0
#define WS_E0    (WS_LOG + NTOK * NE)
#define WS_E1    (WS_E0 + NTOK)
#define WS_G0    (WS_E1 + NTOK)
#define WS_G1    (WS_G0 + NTOK)
#define WS_R1    (WS_G1 + NTOK)
#define WS_P0    (WS_R1 + NTOK)
#define WS_P1    (WS_P0 + NTOK)
#define WS_CNT   (WS_P1 + NTOK)   // 32 ints  (top-1 histogram per (b,e))
#define WS_PROXY (WS_CNT + 32)    // 32 floats (sum of raw_gates per (b,e))
#define WS_Z2    (WS_PROXY + 32)  // 1 float  (sum of z^2)

// ---------------------------------------------------------------- init / zero
__global__ void init_ws_kernel(float* __restrict__ ws) {
    int i = threadIdx.x;
    if (i < 65) ws[WS_CNT + i] = 0.0f;   // zeros CNT(32) + PROXY(32) + Z2(1)
}

__global__ void zero_out_kernel(float4* __restrict__ p, size_t nvec) {
    size_t i = (size_t)blockIdx.x * blockDim.x + threadIdx.x;
    size_t stride = (size_t)gridDim.x * blockDim.x;
    float4 z = make_float4(0.f, 0.f, 0.f, 0.f);
    for (; i < nvec; i += stride) p[i] = z;
}

// ------------------------------------------------------- logits via WMMA f16
// One wave -> 16 tokens x 16 experts (8 real, 8 mirrored/discarded), K=1024.
// Two independent accumulator chains to overlap WMMA latency with loads.
__global__ void __launch_bounds__(128) logits_wmma_kernel(
    const float* __restrict__ x, const float* __restrict__ Wg,
    float* __restrict__ logits)
{
    const int tid  = threadIdx.x;
    const int lane = tid & 31;
    const int wave = (blockIdx.x * blockDim.x + tid) >> 5;   // 0..511
    const int row0 = wave * 16;
    const int half = lane >> 4;          // lane group 0: lanes 0-15, 1: 16-31
    const int row  = row0 + (lane & 15); // A-matrix row for this lane
    const int ncol = lane & 15;          // B/C column for this lane
    // Columns 8..15 mirror rows 0..7 of Wg; their products land in C columns
    // 8..15 which are never stored, so no zero-padding (and no selects) needed.
    const float* wrow = Wg + (size_t)(ncol & 7) * DIM_;
    const float* xrow = x + (size_t)row * DIM_;

    v8f c0 = {}, c1 = {};
    for (int k0 = 0; k0 < DIM_; k0 += 64) {
#pragma unroll
        for (int u = 0; u < 2; ++u) {
            const int kb = k0 + u * 32;
            // A: 16-bit A 16x32 layout. lanes 0-15: K {0..7,16..23}; lanes 16-31: +8
            const float* pA = xrow + kb + half * 8;
            v16h A;
#pragma unroll
            for (int i = 0; i < 8; ++i) A[i]     = (_Float16)pA[i];
#pragma unroll
            for (int i = 0; i < 8; ++i) A[8 + i] = (_Float16)pA[16 + i];

            // B: 32x16, B[k][n] = Wg[n][k]; lanes 0-15: K {0..7,16..23}; lanes 16-31: +8
            const float* pW = wrow + kb + half * 8;
            v16h Bm;
#pragma unroll
            for (int i = 0; i < 8; ++i) Bm[i]     = (_Float16)pW[i];
#pragma unroll
            for (int i = 0; i < 8; ++i) Bm[8 + i] = (_Float16)pW[16 + i];

            if (u == 0)
                c0 = __builtin_amdgcn_wmma_f32_16x16x32_f16(
                         false, A, false, Bm, (short)0, c0, false, false);
            else
                c1 = __builtin_amdgcn_wmma_f32_16x16x32_f16(
                         false, A, false, Bm, (short)0, c1, false, false);
        }
    }
    const v8f c = c0 + c1;

    if (ncol < NE) {
        // C layout: lanes 0-15 -> M=r, lanes 16-31 -> M=8+r, N=lane&15
#pragma unroll
        for (int r = 0; r < 8; ++r) {
            int mrow = row0 + half * 8 + r;
            logits[(size_t)mrow * NE + ncol] = c[r];
        }
    }
}

// ------------------------------------------- softmax / top-2 / routing flags
__global__ void __launch_bounds__(256) gating_kernel(
    const float* __restrict__ logits, const float* __restrict__ probs,
    float* __restrict__ ws)
{
    __shared__ float s_proxy[NE];
    __shared__ int   s_cnt[NE];
    __shared__ float s_z2;
    const int tid = threadIdx.x;
    if (tid < NE) { s_proxy[tid] = 0.f; s_cnt[tid] = 0; }
    if (tid == 0) s_z2 = 0.f;
    __syncthreads();

    const int t = blockIdx.x * blockDim.x + tid;   // token id 0..8191
    float li[NE];
#pragma unroll
    for (int e = 0; e < NE; ++e) li[e] = logits[(size_t)t * NE + e];
    float m = li[0];
#pragma unroll
    for (int e = 1; e < NE; ++e) m = fmaxf(m, li[e]);
    float ex[NE], s = 0.f;
#pragma unroll
    for (int e = 0; e < NE; ++e) { ex[e] = __expf(li[e] - m); s += ex[e]; }
    const float inv = 1.f / s;
    const float z = m + __logf(s);
    atomicAdd(&s_z2, z * z);

    float raw[NE];
#pragma unroll
    for (int e = 0; e < NE; ++e) { raw[e] = ex[e] * inv; atomicAdd(&s_proxy[e], raw[e]); }

    int e0 = 0; float v0 = raw[0];
#pragma unroll
    for (int e = 1; e < NE; ++e) if (raw[e] > v0) { v0 = raw[e]; e0 = e; }
    int e1 = 0; float v1 = -1.f;
#pragma unroll
    for (int e = 0; e < NE; ++e) if (e != e0 && raw[e] > v1) { v1 = raw[e]; e1 = e; }
    atomicAdd(&s_cnt[e0], 1);

    const float denom = fmaxf(v0 + v1, 1e-9f);
    const float g0 = v0 / denom, g1 = v1 / denom;
    const int route1 = (probs[NTOK + t] < g1 * 5.0f) ? 1 : 0;   // g1 / 0.2

    ((int*)ws)[WS_E0 + t] = e0;
    ((int*)ws)[WS_E1 + t] = e1;
    ws[WS_G0 + t] = g0;
    ws[WS_G1 + t] = g1;
    ((int*)ws)[WS_R1 + t] = route1;

    __syncthreads();
    const int b = t >> 11;   // uniform per block (256 | 2048)
    if (tid < NE) {
        atomicAdd(&ws[WS_PROXY + b * NE + tid], s_proxy[tid]);
        atomicAdd((int*)ws + WS_CNT + b * NE + tid, s_cnt[tid]);
    }
    if (tid == 0) atomicAdd(&ws[WS_Z2], s_z2);
}

// --------------------------- wave-ballot exclusive scan + capacity positions
// One wave32 per (b, e) pair: 32 waves total.
__global__ void __launch_bounds__(128) scan_kernel(float* __restrict__ ws)
{
    const int tid  = threadIdx.x;
    const int lane = tid & 31;
    const int w = blockIdx.x * 4 + (tid >> 5);   // 0..31
    const int b = w >> 3, e = w & 7;

    const int* e0 = (const int*)ws + WS_E0;
    const int* e1 = (const int*)ws + WS_E1;
    const int* r1 = (const int*)ws + WS_R1;
    int* p0 = (int*)ws + WS_P0;
    int* p1 = (int*)ws + WS_P1;
    const unsigned lmask = (1u << lane) - 1u;

    int running = 0;
    for (int base = 0; base < N_; base += 32) {
        const int t = b * N_ + base + lane;
        const bool bit = (e0[t] == e);
        const unsigned mb = __builtin_amdgcn_ballot_w32(bit);
        if (bit) p0[t] = running + __popc(mb & lmask);
        running += __popc(mb);
    }
    const int prev = running < CAP ? running : CAP;  // min(count0, cap)
    int run2 = 0;
    for (int base = 0; base < N_; base += 32) {
        const int t = b * N_ + base + lane;
        const bool bit = r1[t] && (e1[t] == e);
        const unsigned mb = __builtin_amdgcn_ballot_w32(bit);
        if (bit) p1[t] = prev + run2 + __popc(mb & lmask);
        run2 += __popc(mb);
    }
}

// ------------------------------------------------- sparse scatter of outputs
__global__ void __launch_bounds__(256) scatter_kernel(
    const float* __restrict__ ws, float* __restrict__ out)
{
    const int t = blockIdx.x * blockDim.x + threadIdx.x;   // token id
    float* dispatch = out;
    float* combine  = out + OUT_MAT;
    const int* e0 = (const int*)ws + WS_E0;
    const int* e1 = (const int*)ws + WS_E1;
    const int* r1 = (const int*)ws + WS_R1;
    const int* p0 = (const int*)ws + WS_P0;
    const int* p1 = (const int*)ws + WS_P1;

    { // k = 0 (always routed; may be capacity-dropped)
        const int a = e0[t], pa = p0[t];
        if (pa < CAP) {
            const size_t idx = ((size_t)t * NE + a) * CAP + pa;
            const float g = ws[WS_G0 + t];
            combine[idx]  = g;
            dispatch[idx] = (g != 0.f) ? 1.f : 0.f;
        }
    }
    if (r1[t]) { // k = 1
        const int a = e1[t], pa = p1[t];
        if (pa < CAP) {
            const size_t idx = ((size_t)t * NE + a) * CAP + pa;
            const float g = ws[WS_G1 + t];
            combine[idx]  = g;
            dispatch[idx] = (g != 0.f) ? 1.f : 0.f;
        }
    }
}

// ------------------------------------------------------------------- losses
__global__ void loss_kernel(const float* __restrict__ ws, float* __restrict__ out)
{
    if (threadIdx.x == 0) {
        float bal = 0.f;
        for (int i = 0; i < 32; ++i) {
            const float proxy_mean = ws[WS_PROXY + i] * (1.f / N_);
            const float dens = (float)((const int*)ws)[WS_CNT + i] * (1.f / N_);
            bal += proxy_mean * dens;
        }
        out[2 * OUT_MAT]     = bal * (64.f / 32.f);        // *NE^2, mean over b*e
        out[2 * OUT_MAT + 1] = ws[WS_Z2] * (1.f / NTOK);   // mean z^2
    }
}

extern "C" void kernel_launch(void* const* d_in, const int* in_sizes, int n_in,
                              void* d_out, int out_size, void* d_ws, size_t ws_size,
                              hipStream_t stream)
{
    (void)in_sizes; (void)n_in; (void)out_size; (void)ws_size;
    const float* x     = (const float*)d_in[0];
    const float* Wg    = (const float*)d_in[1];
    const float* probs = (const float*)d_in[2];
    float* out = (float*)d_out;
    float* ws  = (float*)d_ws;

    hipLaunchKernelGGL(init_ws_kernel, dim3(1), dim3(128), 0, stream, ws);
    hipLaunchKernelGGL(zero_out_kernel, dim3(2048), dim3(256), 0, stream,
                       (float4*)out, (OUT_MAT * 2) / 4);
    hipLaunchKernelGGL(logits_wmma_kernel, dim3(128), dim3(128), 0, stream,
                       x, Wg, ws + WS_LOG);
    hipLaunchKernelGGL(gating_kernel, dim3(NTOK / 256), dim3(256), 0, stream,
                       ws + WS_LOG, probs, ws);
    hipLaunchKernelGGL(scan_kernel, dim3(8), dim3(128), 0, stream, ws);
    hipLaunchKernelGGL(scatter_kernel, dim3(NTOK / 256), dim3(256), 0, stream, ws, out);
    hipLaunchKernelGGL(loss_kernel, dim3(1), dim3(32), 0, stream, ws, out);
}
